// TeacherGNN_13237089206559
// MI455X (gfx1250) — compile-verified, matrix operand order checked by reading, not run on hardware
//
#include <hip/hip_runtime.h>
#include <hip/hip_bf16.h>

typedef __attribute__((ext_vector_type(16))) _Float16 v16h;
typedef __attribute__((ext_vector_type(8)))  _Float16 v8h_t;
typedef __attribute__((ext_vector_type(8)))  float    v8f;

#define BM 256
#define BN 64
#define BK 32
#define NNODES 20000
#define NEDGES 320000
#define NGRAPH 128
#define CHUNK  32000
#define NCHUNK 10

// ---------------------------------------------------------------------------
// CDNA5 async global->LDS staging (ASYNCcnt-tracked, bypasses VGPRs)
// ---------------------------------------------------------------------------
__device__ inline void async_g2l_b128(void* lds, const void* gaddr) {
    asm volatile("global_load_async_to_lds_b128 %0, %1, off"
                 :: "v"((unsigned)(unsigned long long)lds), "v"(gaddr)
                 : "memory");
}
__device__ inline void wait_async0() {
    asm volatile("s_wait_asynccnt 0x0" ::: "memory");
}

// ---------------------------------------------------------------------------
// small prep kernels
// ---------------------------------------------------------------------------
__global__ void zero_f32_kernel(float* p, int n) {
    int i = blockIdx.x * blockDim.x + threadIdx.x;
    if (i < n) p[i] = 0.0f;
}

__global__ void convert_w_kernel(const float* __restrict__ w, _Float16* __restrict__ w16, int n) {
    int i = blockIdx.x * blockDim.x + threadIdx.x;
    if (i < n) w16[i] = (_Float16)w[i];
}

// fold Linear bias + BatchNorm(eval): y = acc*scale + shift
__global__ void bn_prep_kernel(const float* __restrict__ b, const float* __restrict__ g,
                               const float* __restrict__ beta, const float* __restrict__ mu,
                               const float* __restrict__ var,
                               float* __restrict__ scale, float* __restrict__ shift, int n) {
    int i = threadIdx.x;
    if (i < n) {
        float s = g[i] * rsqrtf(var[i] + 1e-5f);
        scale[i] = s;
        shift[i] = (b[i] - mu[i]) * s + beta[i];
    }
}

__global__ void convert_x_kernel(const float* __restrict__ x, _Float16* __restrict__ h, int n) {
    int i = blockIdx.x * blockDim.x + threadIdx.x;
    if (i < n) h[i] = (_Float16)x[i];
}

__global__ void count_edges_kernel(const int* __restrict__ dst, float* __restrict__ cnt, int e) {
    int i = blockIdx.x * blockDim.x + threadIdx.x;
    if (i < e) atomicAdd(&cnt[dst[i]], 1.0f);
}

__global__ void finalize_kernel(const float* __restrict__ sum, const float* __restrict__ cnt,
                                _Float16* __restrict__ out, int total, int dout) {
    int i = blockIdx.x * blockDim.x + threadIdx.x;
    if (i < total) {
        int node = i / dout;
        float c = cnt[node];
        c = c < 1.0f ? 1.0f : c;
        float v = sum[i] / c;
        out[i] = (_Float16)(v > 0.0f ? v : 0.0f);
    }
}

// post-ReLU values are >= 0, so u32 bit-pattern max == float max
__global__ void pool_kernel(const _Float16* __restrict__ h, const int* __restrict__ batch,
                            float* __restrict__ gpool, int nnodes) {
    int i = blockIdx.x * blockDim.x + threadIdx.x;
    if (i < nnodes * 256) {
        int node = i >> 8, n = i & 255;
        float v = (float)h[i];
        if (v < 0.0f) v = 0.0f;
        atomicMax((unsigned int*)&gpool[batch[node] * 256 + n], __float_as_uint(v));
    }
}

__global__ void fc_head_kernel(const float* __restrict__ gpool,
                               const float* __restrict__ W1, const float* __restrict__ b1,
                               const float* __restrict__ Wo, const float* __restrict__ bo,
                               float* __restrict__ out) {
    __shared__ float sg[256];
    __shared__ float sh1[256];
    int g = blockIdx.x, t = threadIdx.x;
    sg[t] = gpool[g * 256 + t];
    __syncthreads();
    float acc = b1[t];
    for (int k = 0; k < 256; ++k) acc += sg[k] * W1[t * 256 + k];
    sh1[t] = acc > 0.0f ? acc : 0.0f;
    __syncthreads();
    if (t < 2) {
        float a = bo[t];
        for (int k = 0; k < 256; ++k) a += sh1[k] * Wo[t * 256 + k];
        out[g * 2 + t] = a;
    }
}

// ---------------------------------------------------------------------------
// WMMA edge-MLP GEMM, fully templated on sizes so address math const-folds:
//   MODE 0: A gathered = [h[dst], h[src]-h[dst]]  -> dense f16 out
//   MODE 1: A dense from mbuf                     -> dense f16 out
//   MODE 2: A dense from mbuf                     -> atomicAdd into nodesum[dst]
// Block: 256 threads = 8 wave32 in 4(M) x 2(N); block tile 256x64, wave 64x32.
// Double-buffered LDS fed by global_load_async_to_lds_b128.
// ---------------------------------------------------------------------------
__device__ inline v16h load_frag(const _Float16* rowbase, int hi) {
    // ISA 16-bit A/B layout: low lanes hold K {0..7,16..23}, high lanes K {8..15,24..31}
    v8h_t lo = *(const v8h_t*)(rowbase + 8 * hi);
    v8h_t hh = *(const v8h_t*)(rowbase + 16 + 8 * hi);
    v16h r;
#pragma unroll
    for (int i = 0; i < 8; ++i) { r[i] = lo[i]; r[i + 8] = hh[i]; }
    return r;
}

template <int MODE, int K, int DIN, int NOUT>
__global__ __launch_bounds__(256)
void edge_gemm_kernel(const _Float16* __restrict__ hin,   // node feats (MODE 0) [NNODES, DIN]
                      const _Float16* __restrict__ ain,   // dense A (MODE 1/2)  [CHUNK, K]
                      const int* __restrict__ esrc, const int* __restrict__ edst,
                      long long e0,
                      const _Float16* __restrict__ w16,   // [NOUT, K]
                      const float* __restrict__ scale, const float* __restrict__ shift,
                      _Float16* __restrict__ mout,        // dense out [CHUNK, NOUT]
                      float* __restrict__ nodesum)        // atomic out [NNODES, NOUT]
{
    constexpr bool SMALLK = (K < BK);
    constexpr int NK = SMALLK ? 1 : (K / BK);

    __shared__ _Float16 As[2][BM * BK];
    __shared__ _Float16 Bs[2][BN * BK];

    const int t = threadIdx.x;
    const int lane = t & 31, wave = t >> 5;
    const int wm = wave & 3, wn = wave >> 2;
    const unsigned mbase = blockIdx.x * BM;               // chunk-local edge row
    const int nbase = blockIdx.y * BN;
    const int hi = lane >> 4, lml = lane & 15;

    v8f acc[4][2] = {};

    // A staging: one row per thread
    const int ar = t;
    const _Float16* hs = nullptr;
    const _Float16* hd = nullptr;
    const _Float16* arow = nullptr;
    if (MODE == 0) {
        long long e = e0 + mbase + ar;
        hs = hin + (unsigned)esrc[e] * DIN;
        hd = hin + (unsigned)edst[e] * DIN;
    } else {
        arow = ain + (mbase + ar) * (unsigned)K;
    }
    // B staging: 4 threads per output channel, 8 halves (16B) each
    const int br = t >> 2;
    const int bp = (t & 3) * 8;
    const _Float16* brow = w16 + (unsigned)(nbase + br) * K + bp;

    // ---- stage one K-tile into LDS buffer `buf`
    auto stage = [&](int k0, int buf) {
        _Float16* aDst = &As[buf][ar * BK];
        if (MODE == 0) {
            if (SMALLK) {
#pragma unroll
                for (int i = 0; i < BK; ++i) {
                    int kg = k0 + i;
                    _Float16 v = (_Float16)0.0f;
                    if (kg < DIN)     v = hd[kg];
                    else if (kg < K)  v = (_Float16)((float)hs[kg - DIN] - (float)hd[kg - DIN]);
                    aDst[i] = v;
                }
            } else if (k0 < DIN) {
                // first half: straight copy of h[dst] row -> async DMA
                const char* g = (const char*)(hd + k0);
#pragma unroll
                for (int i = 0; i < 4; ++i) async_g2l_b128((char*)aDst + 16 * i, g + 16 * i);
            } else {
                // second half: h[src]-h[dst], computed on VALU (packed f16)
                const v8h_t* ps = (const v8h_t*)(hs + (k0 - DIN));
                const v8h_t* pd = (const v8h_t*)(hd + (k0 - DIN));
                v8h_t* dv = (v8h_t*)aDst;
#pragma unroll
                for (int i = 0; i < 4; ++i) dv[i] = ps[i] - pd[i];
            }
        } else {
            const char* g = (const char*)(arow + k0);
#pragma unroll
            for (int i = 0; i < 4; ++i) async_g2l_b128((char*)aDst + 16 * i, g + 16 * i);
        }
        _Float16* bDst = &Bs[buf][br * BK + bp];
        if (SMALLK) {
#pragma unroll
            for (int i = 0; i < 8; ++i)
                bDst[i] = (k0 + bp + i < K) ? brow[k0 + i] : (_Float16)0.0f;
        } else {
            async_g2l_b128(bDst, brow + k0);
        }
    };

    stage(0, 0);
    for (int kt = 0; kt < NK; ++kt) {
        wait_async0();
        __syncthreads();
        if (kt + 1 < NK) stage((kt + 1) * BK, (kt + 1) & 1);

        const int buf = kt & 1;
        v16h af[4], bf[2];
#pragma unroll
        for (int mt = 0; mt < 4; ++mt)
            af[mt] = load_frag(&As[buf][(wm * 64 + mt * 16 + lml) * BK], hi);
#pragma unroll
        for (int nt = 0; nt < 2; ++nt)
            bf[nt] = load_frag(&Bs[buf][(wn * 32 + nt * 16 + lml) * BK], hi);
#pragma unroll
        for (int mt = 0; mt < 4; ++mt)
#pragma unroll
            for (int nt = 0; nt < 2; ++nt)
                acc[mt][nt] = __builtin_amdgcn_wmma_f32_16x16x32_f16(
                    false, af[mt], false, bf[nt], (short)0, acc[mt][nt], false, false);
    }

    // ---- epilogue: BN fold + ReLU, then f16 store or f32 scatter-add
    // all indices 32-bit; NOUT constexpr so per-row deltas fold into store offsets
#pragma unroll
    for (int nt = 0; nt < 2; ++nt) {
        const int ng = nbase + wn * 32 + nt * 16 + lml;
        const float sc = scale[ng], sh = shift[ng];
#pragma unroll
        for (int mt = 0; mt < 4; ++mt) {
            const unsigned erow = mbase + wm * 64 + mt * 16 + 8 * hi;  // chunk-local base row
            if (MODE == 2) {
                const int* eptr = edst + e0 + erow;
#pragma unroll
                for (int r = 0; r < 8; ++r) {
                    float y = acc[mt][nt][r] * sc + sh;
                    y = y > 0.0f ? y : 0.0f;
                    atomicAdd(&nodesum[(unsigned)eptr[r] * NOUT + ng], y);
                }
            } else {
                _Float16* prow = mout + erow * NOUT + ng;
#pragma unroll
                for (int r = 0; r < 8; ++r) {
                    float y = acc[mt][nt][r] * sc + sh;
                    y = y > 0.0f ? y : 0.0f;
                    prow[r * NOUT] = (_Float16)y;
                }
            }
        }
    }
}

// ---------------------------------------------------------------------------
// host orchestration
// ---------------------------------------------------------------------------
extern "C" void kernel_launch(void* const* d_in, const int* in_sizes, int n_in,
                              void* d_out, int out_size, void* d_ws, size_t ws_size,
                              hipStream_t stream) {
    const float* x   = (const float*)d_in[0];
    const int* ei    = (const int*)d_in[1];
    const int* batch = (const int*)d_in[2];
    const int* esrc = ei;
    const int* edst = ei + NEDGES;

    const int dins[3] = {4, 64, 128};
    const int hsz[3]  = {64, 128, 256};

    char* base = (char*)d_ws;
    size_t off = 0;
    auto alloc = [&](size_t bytes) -> void* {
        void* p = base + off;
        off = (off + bytes + 255) & ~(size_t)255;
        return p;
    };
    _Float16* w16[9]; float* scp[9]; float* shp[9];
    for (int c = 0; c < 3; ++c)
        for (int l = 0; l < 3; ++l) {
            int K = (l == 0) ? 2 * dins[c] : hsz[c];
            w16[c * 3 + l] = (_Float16*)alloc((size_t)hsz[c] * K * 2);
            scp[c * 3 + l] = (float*)alloc((size_t)hsz[c] * 4);
            shp[c * 3 + l] = (float*)alloc((size_t)hsz[c] * 4);
        }
    _Float16* hA = (_Float16*)alloc((size_t)NNODES * 256 * 2);
    _Float16* hB = (_Float16*)alloc((size_t)NNODES * 256 * 2);
    float* nodesum = (float*)alloc((size_t)NNODES * 256 * 4);
    float* cnt     = (float*)alloc((size_t)NNODES * 4);
    _Float16* mb0  = (_Float16*)alloc((size_t)CHUNK * 256 * 2);
    _Float16* mb1  = (_Float16*)alloc((size_t)CHUNK * 256 * 2);
    float* gpool   = (float*)alloc((size_t)NGRAPH * 256 * 4);

    // weight convert + BN fold (params flattened: W,b,gamma,beta,mean,var per layer)
    for (int c = 0; c < 3; ++c)
        for (int l = 0; l < 3; ++l) {
            int pb = 3 + c * 18 + l * 6;
            int K = (l == 0) ? 2 * dins[c] : hsz[c];
            int n = hsz[c], wc = n * K;
            convert_w_kernel<<<(wc + 255) / 256, 256, 0, stream>>>((const float*)d_in[pb], w16[c * 3 + l], wc);
            bn_prep_kernel<<<1, n, 0, stream>>>(
                (const float*)d_in[pb + 1], (const float*)d_in[pb + 2], (const float*)d_in[pb + 3],
                (const float*)d_in[pb + 4], (const float*)d_in[pb + 5],
                scp[c * 3 + l], shp[c * 3 + l], n);
        }

    zero_f32_kernel<<<(NNODES + 255) / 256, 256, 0, stream>>>(cnt, NNODES);
    count_edges_kernel<<<(NEDGES + 255) / 256, 256, 0, stream>>>(edst, cnt, NEDGES);
    convert_x_kernel<<<(NNODES * 4 + 255) / 256, 256, 0, stream>>>(x, hA, NNODES * 4);

    _Float16* hcur = hA;
    _Float16* hnext = hB;
    for (int c = 0; c < 3; ++c) {
        int h = hsz[c];
        zero_f32_kernel<<<(NNODES * h + 255) / 256, 256, 0, stream>>>(nodesum, NNODES * h);
        dim3 gg(CHUNK / BM, h / BN);
        for (int ch = 0; ch < NCHUNK; ++ch) {
            long long e0 = (long long)ch * CHUNK;
            if (c == 0) {
                edge_gemm_kernel<0,   8,   4,  64><<<gg, 256, 0, stream>>>(hcur, nullptr, esrc, edst, e0,
                    w16[0], scp[0], shp[0], mb0, nullptr);
                edge_gemm_kernel<1,  64,   0,  64><<<gg, 256, 0, stream>>>(nullptr, mb0, esrc, edst, e0,
                    w16[1], scp[1], shp[1], mb1, nullptr);
                edge_gemm_kernel<2,  64,   0,  64><<<gg, 256, 0, stream>>>(nullptr, mb1, esrc, edst, e0,
                    w16[2], scp[2], shp[2], nullptr, nodesum);
            } else if (c == 1) {
                edge_gemm_kernel<0, 128,  64, 128><<<gg, 256, 0, stream>>>(hcur, nullptr, esrc, edst, e0,
                    w16[3], scp[3], shp[3], mb0, nullptr);
                edge_gemm_kernel<1, 128,   0, 128><<<gg, 256, 0, stream>>>(nullptr, mb0, esrc, edst, e0,
                    w16[4], scp[4], shp[4], mb1, nullptr);
                edge_gemm_kernel<2, 128,   0, 128><<<gg, 256, 0, stream>>>(nullptr, mb1, esrc, edst, e0,
                    w16[5], scp[5], shp[5], nullptr, nodesum);
            } else {
                edge_gemm_kernel<0, 256, 128, 256><<<gg, 256, 0, stream>>>(hcur, nullptr, esrc, edst, e0,
                    w16[6], scp[6], shp[6], mb0, nullptr);
                edge_gemm_kernel<1, 256,   0, 256><<<gg, 256, 0, stream>>>(nullptr, mb0, esrc, edst, e0,
                    w16[7], scp[7], shp[7], mb1, nullptr);
                edge_gemm_kernel<2, 256,   0, 256><<<gg, 256, 0, stream>>>(nullptr, mb1, esrc, edst, e0,
                    w16[8], scp[8], shp[8], nullptr, nodesum);
            }
        }
        finalize_kernel<<<(NNODES * h + 255) / 256, 256, 0, stream>>>(nodesum, cnt, hnext, NNODES * h, h);
        _Float16* tmp = hcur; hcur = hnext; hnext = tmp;
    }

    zero_f32_kernel<<<(NGRAPH * 256 + 255) / 256, 256, 0, stream>>>(gpool, NGRAPH * 256);
    pool_kernel<<<(NNODES * 256 + 255) / 256, 256, 0, stream>>>(hcur, batch, gpool, NNODES);
    fc_head_kernel<<<NGRAPH, 256, 0, stream>>>(gpool,
        (const float*)d_in[57], (const float*)d_in[58],
        (const float*)d_in[59], (const float*)d_in[60], (float*)d_out);
}
